// KSpaceTransformerGNNEncoder_7653631721504
// MI455X (gfx1250) — matrix operand order repaired and checked by reading, not compiled, not run on hardware
//
#include <hip/hip_runtime.h>
#include <cstdint>
#include <cstddef>

namespace {

constexpr int   kN     = 50000;
constexpr int   kE     = 800000;
constexpr int   kFin   = 16;
constexpr int   kHid   = 32;
constexpr int   kHeads = 8;
constexpr int   kCout  = 128;
constexpr int   kB     = 32;
constexpr int   kD     = 256;      // HID*HEADS
constexpr float kEps   = 1e-5f;
constexpr float kScale = 0.17677669529663687f;  // 1/sqrt(32)

typedef __attribute__((ext_vector_type(16))) __bf16 v16bf;
typedef __attribute__((ext_vector_type(8)))  float  v8f;

__device__ __forceinline__ __bf16 f2bf(float x) {
  unsigned int u = __float_as_uint(x);
  u = (u + 0x7FFFu + ((u >> 16) & 1u)) >> 16;   // round-to-nearest-even
  unsigned short s = (unsigned short)u;
  __bf16 r;
  __builtin_memcpy(&r, &s, 2);
  return r;
}

// order-preserving float<->uint map for atomicMax-based segment max
__device__ __forceinline__ unsigned int ford(float x) {
  unsigned int u = __float_as_uint(x);
  return (u & 0x80000000u) ? ~u : (u | 0x80000000u);
}
__device__ __forceinline__ float fdec(unsigned int u) {
  return (u & 0x80000000u) ? __uint_as_float(u & 0x7FFFFFFFu)
                           : __uint_as_float(~u);
}

// ---------------- initial projection: h = x @ w_init + b_init  [N,32] -----
__global__ void k_init_proj(const float* __restrict__ x,
                            const float* __restrict__ w,
                            const float* __restrict__ b,
                            float* __restrict__ h) {
  int idx = blockIdx.x * blockDim.x + threadIdx.x;
  if (idx >= kN * kHid) return;
  int n = idx >> 5, c = idx & 31;
  float acc = b[c];
  const float* xr = x + (size_t)n * kFin;
#pragma unroll
  for (int k = 0; k < kFin; ++k) acc += xr[k] * w[k * kHid + c];
  h[idx] = acc;
}

// ---------------- pack f32 -> bf16 in WMMA fragment order -----------------
// CDNA5 wave32 fragment layouts per CDNA5 ISA:
//   A (16x32 bf16): M = lane%16, K(j) = (j/8)*16 + (lane/16)*8 + j%8
//   B (32x16 bf16): N = lane%16, K(j) = (j/8)*16 + (lane/16)*8 + j%8
// Packed storage: frag[tile][ktile][lane][j], 512 bf16 per (tile,ktile) slab.
__global__ void k_pack_a(const float* __restrict__ src, __bf16* __restrict__ dst,
                         int d, int count) {   // count = nrows*d
  int idx = blockIdx.x * blockDim.x + threadIdx.x;
  if (idx >= count) return;
  int j  = idx & 15;
  int L  = (idx >> 4) & 31;
  int t  = idx >> 9;
  int KT = d >> 5;
  int kt = t % KT;
  int rt = t / KT;
  int M = rt * 16 + (L & 15);
  int K = kt * 32 + ((j >> 3) << 4) + ((L >> 4) << 3) + (j & 7);
  dst[idx] = f2bf(src[(size_t)M * d + K]);
}

__global__ void k_pack_b(const float* __restrict__ src, __bf16* __restrict__ dst,
                         int d) {              // src = W[d,256]; count = d*256
  int idx = blockIdx.x * blockDim.x + threadIdx.x;
  if (idx >= d * kD) return;
  int j  = idx & 15;
  int L  = (idx >> 4) & 31;
  int t  = idx >> 9;
  int KT = d >> 5;
  int kt = t % KT;
  int ct = t / KT;
  int Nc = ct * 16 + (L & 15);
  int K  = kt * 32 + ((j >> 3) << 4) + ((L >> 4) << 3) + (j & 7);
  dst[idx] = f2bf(src[(size_t)K * kD + Nc]);
}

// ---------------- WMMA GEMM:  Y[N,256] = A[N,d] @ W[d,256] + bias ---------
// One wave per 16x32 output tile (two 16x16 accumulators sharing one A frag).
// Operands pre-swizzled: each lane reads contiguous 32B per fragment
// (2x global_load_b128), fully coalesced across the wave.
__global__ __launch_bounds__(256) void k_gemm_wmma(
    const __bf16* __restrict__ Asw, const __bf16* __restrict__ Bsw,
    const float* __restrict__ bias, float* __restrict__ Y,
    int nrt, int KT) {
  const int wave = threadIdx.x >> 5;
  const int lane = threadIdx.x & 31;
  const int rt   = blockIdx.x * 8 + wave;
  if (rt >= nrt) return;                    // wave-uniform; EXEC stays all-1s
  const int ct0 = blockIdx.y * 2;

  const __bf16* ap  = Asw + (size_t)rt * KT * 512 + lane * 16;
  const __bf16* bp0 = Bsw + (size_t)ct0 * KT * 512 + lane * 16;
  const __bf16* bp1 = bp0 + (size_t)KT * 512;

  v8f acc0 = {}, acc1 = {};
  for (int kt = 0; kt < KT; ++kt) {
    v16bf a  = *reinterpret_cast<const v16bf*>(ap  + (size_t)kt * 512);
    v16bf b0 = *reinterpret_cast<const v16bf*>(bp0 + (size_t)kt * 512);
    v16bf b1 = *reinterpret_cast<const v16bf*>(bp1 + (size_t)kt * 512);
    acc0 = __builtin_amdgcn_wmma_f32_16x16x32_bf16(false, a, false, b0,
                                                   (short)0, acc0, false, false);
    acc1 = __builtin_amdgcn_wmma_f32_16x16x32_bf16(false, a, false, b1,
                                                   (short)0, acc1, false, false);
  }

  const int half = lane >> 4, l16 = lane & 15;
  float bv0 = bias[ct0 * 16 + l16];
  float bv1 = bias[ct0 * 16 + 16 + l16];
#pragma unroll
  for (int r = 0; r < 8; ++r) {
    int row = rt * 16 + r + half * 8;
    Y[(size_t)row * kD + ct0 * 16 + l16]      = acc0[r] + bv0;
    Y[(size_t)row * kD + ct0 * 16 + 16 + l16] = acc1[r] + bv1;
  }
}

// ---------------- per-edge attention logits + segment max -----------------
__global__ void k_logits(const int* __restrict__ srcv, const int* __restrict__ dstv,
                         const float* __restrict__ q, const float* __restrict__ kx,
                         float* __restrict__ logits, unsigned int* __restrict__ menc) {
  int idx = blockIdx.x * blockDim.x + threadIdx.x;
  if (idx >= kE * kHeads) return;
  int e = idx >> 3, hh = idx & 7;
  int s = srcv[e], dd = dstv[e];
  const float4* qp = reinterpret_cast<const float4*>(q + (size_t)dd * kD + hh * kHid);
  const float4* kp = reinterpret_cast<const float4*>(kx + (size_t)s * kD + hh * kHid);
  float dot = 0.f;
#pragma unroll
  for (int i = 0; i < 8; ++i) {
    float4 a = qp[i], b = kp[i];
    dot += a.x * b.x + a.y * b.y + a.z * b.z + a.w * b.w;
  }
  float lg = dot * kScale;
  logits[idx] = lg;
  atomicMax(&menc[dd * kHeads + hh], ford(lg));
}

// ---------------- exp(logit - max) and segment sum (in-place) -------------
__global__ void k_expsum(const int* __restrict__ dstv, float* __restrict__ logits,
                         const unsigned int* __restrict__ menc,
                         float* __restrict__ ssum) {
  int idx = blockIdx.x * blockDim.x + threadIdx.x;
  if (idx >= kE * kHeads) return;
  int e = idx >> 3, hh = idx & 7;
  int dd = dstv[e];
  float m  = fdec(menc[dd * kHeads + hh]);
  float ee = __expf(logits[idx] - m);
  logits[idx] = ee;
  atomicAdd(&ssum[dd * kHeads + hh], ee);
}

// ---------------- message scatter: out[dst] += v[src] * alpha -------------
__global__ void k_message(const int* __restrict__ srcv, const int* __restrict__ dstv,
                          const float* __restrict__ v, const float* __restrict__ eexp,
                          const float* __restrict__ ssum, float* __restrict__ out) {
  long long idx = (long long)blockIdx.x * blockDim.x + threadIdx.x;
  if (idx >= (long long)kE * 64) return;
  int e  = (int)(idx >> 6);
  int r  = (int)(idx & 63);
  int hh = r >> 3;
  int c  = (r & 7) << 2;
  int s = srcv[e], dd = dstv[e];
  float alpha = eexp[e * kHeads + hh] / ssum[dd * kHeads + hh];
  float4 vv = *reinterpret_cast<const float4*>(v + (size_t)s * kD + hh * kHid + c);
  float* op = out + (size_t)dd * kD + hh * kHid + c;
  atomicAdd(op + 0, vv.x * alpha);
  atomicAdd(op + 1, vv.y * alpha);
  atomicAdd(op + 2, vv.z * alpha);
  atomicAdd(op + 3, vv.w * alpha);
}

// ---------------- beta-gated combine (one wave32 per node) ----------------
__global__ __launch_bounds__(256) void k_combine(
    const float* __restrict__ skip, const float* __restrict__ out,
    const float* __restrict__ wbeta, float* __restrict__ h) {
  int wave = threadIdx.x >> 5;
  int lane = threadIdx.x & 31;
  int node = blockIdx.x * 8 + wave;
  if (node >= kN) return;
  const float* sp = skip + (size_t)node * kD + lane * 8;
  const float* op = out  + (size_t)node * kD + lane * 8;
  float sk[8], ot[8];
  float p = 0.f;
#pragma unroll
  for (int t = 0; t < 8; ++t) {
    int c = lane * 8 + t;
    float s = sp[t], o = op[t];
    sk[t] = s; ot[t] = o;
    p += s * wbeta[c] + o * wbeta[kD + c] + (s - o) * wbeta[2 * kD + c];
  }
#pragma unroll
  for (int off = 16; off > 0; off >>= 1) p += __shfl_xor(p, off, 32);
  float beta = 1.f / (1.f + __expf(-p));
  float* hp = h + (size_t)node * kD + lane * 8;
#pragma unroll
  for (int t = 0; t < 8; ++t) hp[t] = beta * sk[t] + (1.f - beta) * ot[t];
}

// ---------------- batch-norm stats (per-channel sum / sumsq) --------------
__global__ __launch_bounds__(256) void k_bnstats(const float* __restrict__ h,
                                                 float* __restrict__ bsum,
                                                 float* __restrict__ bsq) {
  int c  = threadIdx.x;
  int r0 = blockIdx.x * 256;
  int r1 = r0 + 256 < kN ? r0 + 256 : kN;
  float s = 0.f, q = 0.f;
  for (int r = r0; r < r1; ++r) {
    float v = h[(size_t)r * kD + c];
    s += v; q += v * v;
  }
  atomicAdd(&bsum[c], s);
  atomicAdd(&bsq[c], q);
}

__global__ void k_bnapply(float* __restrict__ h, const float* __restrict__ bsum,
                          const float* __restrict__ bsq,
                          const float* __restrict__ gamma,
                          const float* __restrict__ beta) {
  int idx = blockIdx.x * blockDim.x + threadIdx.x;
  if (idx >= kN * kD) return;
  int c = idx & 255;
  float mu  = bsum[c] * (1.f / kN);
  float var = bsq[c] * (1.f / kN) - mu * mu;
  float v = (h[idx] - mu) * rsqrtf(var + kEps) * gamma[c] + beta[c];
  h[idx] = v > 0.f ? v : 0.f;
}

// ---------------- graph pooling + final projection ------------------------
__global__ void k_pool(const float* __restrict__ h, const int* __restrict__ batch,
                       float* __restrict__ pooled, float* __restrict__ cnt) {
  int idx = blockIdx.x * blockDim.x + threadIdx.x;
  if (idx >= kN * kD) return;
  int n = idx >> 8, c = idx & 255;
  int b = batch[n];
  atomicAdd(&pooled[b * kD + c], h[idx]);
  if (c == 0) atomicAdd(&cnt[b], 1.0f);
}

__global__ void k_final(const float* __restrict__ pooled, const float* __restrict__ cnt,
                        const float* __restrict__ wf, const float* __restrict__ bf,
                        float* __restrict__ out) {
  int idx = blockIdx.x * blockDim.x + threadIdx.x;
  if (idx >= kB * kCout) return;
  int b = idx / kCout, o = idx % kCout;
  float inv = 1.f / fmaxf(cnt[b], 1.f);
  float acc = 0.f;
  for (int k = 0; k < kD; ++k) acc += pooled[b * kD + k] * wf[k * kCout + o];
  out[idx] = acc * inv + bf[o];
}

}  // namespace

extern "C" void kernel_launch(void* const* d_in, const int* in_sizes, int n_in,
                              void* d_out, int out_size, void* d_ws, size_t ws_size,
                              hipStream_t stream) {
  (void)n_in; (void)out_size; (void)ws_size;
  // ---- parse inputs in setup_inputs() dict order; lists may arrive as 4
  // separate buffers or one concatenated buffer (disambiguate via in_sizes).
  int cur = 0;
  auto nf = [&]() { return (const float*)d_in[cur++]; };
  auto nlist = [&](const float** dstp, long e0, long e1, long e2, long e3) {
    long tot = e0 + e1 + e2 + e3;
    if ((long)in_sizes[cur] == tot) {
      const float* base = (const float*)d_in[cur++];
      dstp[0] = base;
      dstp[1] = base + e0;
      dstp[2] = base + e0 + e1;
      dstp[3] = base + e0 + e1 + e2;
    } else {
      for (int i = 0; i < 4; ++i) dstp[i] = (const float*)d_in[cur++];
    }
  };

  const float* x      = nf();
  const int*   edge   = (const int*)d_in[cur++];
  const int*   batch  = (const int*)d_in[cur++];
  const float* w_init = nf();
  const float* b_init = nf();
  const long W0 = (long)kHid * kD, W1 = (long)kD * kD;
  const float *wq[4], *wk[4], *wv[4], *bq[4], *bk[4], *bv[4];
  const float *wskip[4], *bskip[4], *wbeta[4], *gam[4], *bet[4];
  nlist(wq, W0, W1, W1, W1);
  nlist(wk, W0, W1, W1, W1);
  nlist(wv, W0, W1, W1, W1);
  nlist(bq, kD, kD, kD, kD);
  nlist(bk, kD, kD, kD, kD);
  nlist(bv, kD, kD, kD, kD);
  nlist(wskip, W0, W1, W1, W1);
  nlist(bskip, kD, kD, kD, kD);
  nlist(wbeta, 3 * kD, 3 * kD, 3 * kD, 3 * kD);
  nlist(gam, kD, kD, kD, kD);
  nlist(bet, kD, kD, kD, kD);
  const float* w_final = nf();
  const float* b_final = nf();

  // ---- carve workspace ---------------------------------------------------
  char* p = (char*)d_ws;
  auto carve = [&](size_t bytes) -> void* {
    void* r = (void*)p;
    p += (bytes + 255) & ~(size_t)255;
    return r;
  };
  float*        h    = (float*)carve((size_t)kN * kD * 4);
  __bf16*       hb   = (__bf16*)carve((size_t)kN * kD * 2);   // A, packed
  float*        bufQ = (float*)carve((size_t)kN * kD * 4);    // q, later v
  float*        bufK = (float*)carve((size_t)kN * kD * 4);    // k, later skip
  float*        bufO = (float*)carve((size_t)kN * kD * 4);    // aggregated out
  __bf16*       Wb   = (__bf16*)carve((size_t)kD * kD * 2);   // B, packed
  float*        elog = (float*)carve((size_t)kE * kHeads * 4);  // logits -> exp
  unsigned int* menc = (unsigned int*)carve((size_t)kN * kHeads * 4);
  float*        ssum = (float*)carve((size_t)kN * kHeads * 4);
  float*        bsum = (float*)carve(kD * 4);
  float*        bsq  = (float*)carve(kD * 4);
  float*        pool = (float*)carve((size_t)kB * kD * 4);
  float*        cnt  = (float*)carve(kB * 4);

  const int* srcv = edge;
  const int* dstv = edge + kE;
  dim3 blk(256);
  const int nrt = kN / 16;                      // 3125 row tiles

  k_init_proj<<<(kN * kHid + 255) / 256, blk, 0, stream>>>(x, w_init, b_init, h);

  for (int L = 0; L < 4; ++L) {
    int d    = (L == 0) ? kHid : kD;
    int KT   = d / 32;
    int acnt = kN * d;
    int wcnt = d * kD;
    dim3 ggrid((nrt + 7) / 8, kD / 32);         // 8 waves/block, 2 col-tiles/wave

    k_pack_a<<<(acnt + 255) / 256, blk, 0, stream>>>(h, hb, d, acnt);

    // Q and K projections (WMMA)
    k_pack_b<<<(wcnt + 255) / 256, blk, 0, stream>>>(wq[L], Wb, d);
    k_gemm_wmma<<<ggrid, blk, 0, stream>>>(hb, Wb, bq[L], bufQ, nrt, KT);
    k_pack_b<<<(wcnt + 255) / 256, blk, 0, stream>>>(wk[L], Wb, d);
    k_gemm_wmma<<<ggrid, blk, 0, stream>>>(hb, Wb, bk[L], bufK, nrt, KT);

    // segment softmax over incoming edges
    hipMemsetAsync(menc, 0, (size_t)kN * kHeads * 4, stream);
    hipMemsetAsync(ssum, 0, (size_t)kN * kHeads * 4, stream);
    k_logits<<<(kE * kHeads + 255) / 256, blk, 0, stream>>>(srcv, dstv, bufQ, bufK,
                                                            elog, menc);
    k_expsum<<<(kE * kHeads + 255) / 256, blk, 0, stream>>>(dstv, elog, menc, ssum);

    // V (reuse bufQ) and skip (reuse bufK) projections (WMMA)
    k_pack_b<<<(wcnt + 255) / 256, blk, 0, stream>>>(wv[L], Wb, d);
    k_gemm_wmma<<<ggrid, blk, 0, stream>>>(hb, Wb, bv[L], bufQ, nrt, KT);
    k_pack_b<<<(wcnt + 255) / 256, blk, 0, stream>>>(wskip[L], Wb, d);
    k_gemm_wmma<<<ggrid, blk, 0, stream>>>(hb, Wb, bskip[L], bufK, nrt, KT);

    // weighted message aggregation
    hipMemsetAsync(bufO, 0, (size_t)kN * kD * 4, stream);
    long long mth = (long long)kE * 64;
    k_message<<<(unsigned)((mth + 255) / 256), blk, 0, stream>>>(srcv, dstv, bufQ,
                                                                 elog, ssum, bufO);

    // gated combine -> h, then batch-norm + ReLU in place
    k_combine<<<(kN + 7) / 8, blk, 0, stream>>>(bufK, bufO, wbeta[L], h);
    hipMemsetAsync(bsum, 0, kD * 4, stream);
    hipMemsetAsync(bsq, 0, kD * 4, stream);
    k_bnstats<<<(kN + 255) / 256, blk, 0, stream>>>(h, bsum, bsq);
    k_bnapply<<<(kN * kD + 255) / 256, blk, 0, stream>>>(h, bsum, bsq, gam[L], bet[L]);
  }

  hipMemsetAsync(pool, 0, (size_t)kB * kD * 4, stream);
  hipMemsetAsync(cnt, 0, kB * 4, stream);
  k_pool<<<(kN * kD + 255) / 256, blk, 0, stream>>>(h, batch, pool, cnt);
  k_final<<<(kB * kCout + 255) / 256, blk, 0, stream>>>(pool, cnt, w_final, b_final,
                                                        (float*)d_out);
}
